// Model_24180665876769
// MI455X (gfx1250) — matrix operand order repaired and checked by reading, not compiled, not run on hardware
//
#include <hip/hip_runtime.h>
#include <hip/hip_bf16.h>

typedef _Float16 half_t;
typedef __attribute__((ext_vector_type(16))) _Float16 v16h;
typedef __attribute__((ext_vector_type(8)))  _Float16 v8h;
typedef __attribute__((ext_vector_type(4)))  _Float16 v4h;
typedef __attribute__((ext_vector_type(8)))  float    v8f;

#define TILE_M 128
#define TILE_N 128
#define TILE_K 32
#define LDS_STRIDE 40   // halfs: 80B row stride -> 16B aligned, bank-conflict friendly

__device__ __forceinline__ int imin(int a, int b) { return a < b ? a : b; }

// ---------------------------------------------------------------------------
// Generic f16 WMMA GEMM:  C(MxN) = A(MxK) * Bt(NxK)^T  [+bias][+res][gelu]
// Bt is the WEIGHT-NATIVE layout (out_features x in_features, row-major), so
// model weights need no transpose -- both A and B stage with contiguous-K
// vector loads and ds_store_b128.
// REQUIRES M % 128 == 0 (all call sites satisfy this). K/N arbitrary.
// Double-buffered LDS, register-staged global loads, branchless tails.
// Optional batching via blockIdx.z. Writes fp32 C and/or f16 C.
// ---------------------------------------------------------------------------
__global__ __launch_bounds__(256)
void gemm_f16_wmma(const half_t* __restrict__ A, int ldA, long sA,
                   const half_t* __restrict__ Bt, int ldB, long sB,
                   float* __restrict__ Cf, half_t* __restrict__ Ch, int ldC, long sC,
                   const float* __restrict__ bias,
                   const float* __restrict__ res, int ldR,
                   int M, int N, int K, int doGelu)
{
  __shared__ half_t Als[2][TILE_M * LDS_STRIDE];
  __shared__ half_t Bls[2][TILE_N * LDS_STRIDE];

  const int z = blockIdx.z;
  A  += (size_t)z * sA;
  Bt += (size_t)z * sB;
  if (Cf) Cf += (size_t)z * sC;
  if (Ch) Ch += (size_t)z * sC;

  const int tid   = threadIdx.x;
  const int lane  = tid & 31;
  const int w     = tid >> 5;
  const int waveM = w >> 1;        // 0..3
  const int waveN = w & 1;         // 0..1
  const int mlane = lane & 15;
  const int hh    = lane >> 4;     // lane half

  // loader mapping (identical for A and B: 2 threads per row, 16 halfs each)
  const int lrow = tid >> 1;            // 0..127
  const int lcol = (tid & 1) << 4;      // 0 / 16

  const int gRowA = blockIdx.x * TILE_M + lrow;               // < M by contract
  const int gRowB = imin(blockIdx.y * TILE_N + lrow, N - 1);  // clamp: n>=N cols discarded

  half_t aReg[16], bReg[16];

  auto loadA = [&](int kk) {
    const half_t* ap = A + (size_t)gRowA * ldA + kk + lcol;
    if (kk + lcol + 16 <= K) {
      v8h u0 = *(const v8h*)ap;
      v8h u1 = *(const v8h*)(ap + 8);
      #pragma unroll
      for (int j = 0; j < 8; ++j) { aReg[j] = u0[j]; aReg[j + 8] = u1[j]; }
    } else {
      const half_t* row = A + (size_t)gRowA * ldA;
      #pragma unroll
      for (int j = 0; j < 16; ++j) {
        const int k = kk + lcol + j;
        const int kc = k < K ? k : 0;          // clamped, always valid
        half_t v = row[kc];
        aReg[j] = (k < K) ? v : (half_t)0.f;   // select-zero, no branch
      }
    }
  };
  auto loadB = [&](int kk) {
    const half_t* bp = Bt + (size_t)gRowB * ldB + kk + lcol;
    if (kk + lcol + 16 <= K) {
      v8h u0 = *(const v8h*)bp;
      v8h u1 = *(const v8h*)(bp + 8);
      #pragma unroll
      for (int j = 0; j < 8; ++j) { bReg[j] = u0[j]; bReg[j + 8] = u1[j]; }
    } else {
      const half_t* row = Bt + (size_t)gRowB * ldB;
      #pragma unroll
      for (int j = 0; j < 16; ++j) {
        const int k = kk + lcol + j;
        const int kc = k < K ? k : 0;
        half_t v = row[kc];
        bReg[j] = (k < K) ? v : (half_t)0.f;
      }
    }
  };
  auto storeA = [&](int buf) {
    half_t* dst = &Als[buf][lrow * LDS_STRIDE + lcol];
    v8h u0, u1;
    #pragma unroll
    for (int j = 0; j < 8; ++j) { u0[j] = aReg[j]; u1[j] = aReg[j + 8]; }
    *(v8h*)dst       = u0;
    *(v8h*)(dst + 8) = u1;
  };
  auto storeB = [&](int buf) {
    half_t* dst = &Bls[buf][lrow * LDS_STRIDE + lcol];
    v8h u0, u1;
    #pragma unroll
    for (int j = 0; j < 8; ++j) { u0[j] = bReg[j]; u1[j] = bReg[j + 8]; }
    *(v8h*)dst       = u0;
    *(v8h*)(dst + 8) = u1;
  };

  v8f acc[2][4];
  #pragma unroll
  for (int i = 0; i < 2; ++i)
    #pragma unroll
    for (int j = 0; j < 4; ++j)
      #pragma unroll
      for (int r = 0; r < 8; ++r) acc[i][j][r] = 0.0f;

  const int nIter = (K + TILE_K - 1) / TILE_K;

  loadA(0); loadB(0);
  storeA(0); storeB(0);
  __syncthreads();

  for (int it = 0; it < nIter; ++it) {
    const int cur = it & 1;
    const int kk  = it * TILE_K;

    if (it + 1 < nIter) {
      loadA(kk + TILE_K);           // vmem for next tile overlaps this tile's math
      loadB(kk + TILE_K);
      if (it + 2 < nIter) {         // gfx1250 global_prefetch_b8 two tiles ahead
        const int kpf = imin(kk + 2 * TILE_K + lcol, K - 1);
        __builtin_prefetch(A  + (size_t)gRowA * ldA + kpf, 0, 1);
        __builtin_prefetch(Bt + (size_t)gRowB * ldB + kpf, 0, 1);
      }
    }

    // ---- fragments per CDNA5 ISA layouts (CDNA5 ISA 7.12.2) ----
    v16h afrag[2];
    #pragma unroll
    for (int tm = 0; tm < 2; ++tm) {
      const int r = waveM * 32 + tm * 16 + mlane;  // M row
      v8h lo = *(const v8h*)&Als[cur][r * LDS_STRIDE + 8 * hh];       // K = 8h..8h+7
      v8h hi = *(const v8h*)&Als[cur][r * LDS_STRIDE + 16 + 8 * hh];  // K = 16+8h..
      #pragma unroll
      for (int j = 0; j < 8; ++j) { afrag[tm][j] = lo[j]; afrag[tm][j + 8] = hi[j]; }
    }
    v16h bfrag[4];
    #pragma unroll
    for (int tn = 0; tn < 4; ++tn) {
      const int c = waveN * 64 + tn * 16 + mlane;  // N col
      v8h b0 = *(const v8h*)&Bls[cur][c * LDS_STRIDE + 8 * hh];       // K = 8h..8h+7
      v8h b1 = *(const v8h*)&Bls[cur][c * LDS_STRIDE + 16 + 8 * hh];  // K = 16+8h..
      #pragma unroll
      for (int j = 0; j < 8; ++j) { bfrag[tn][j] = b0[j]; bfrag[tn][j + 8] = b1[j]; }
    }
    #pragma unroll
    for (int tm = 0; tm < 2; ++tm)
      #pragma unroll
      for (int tn = 0; tn < 4; ++tn)
        acc[tm][tn] = __builtin_amdgcn_wmma_f32_16x16x32_f16(
            false, afrag[tm], false, bfrag[tn], (short)0, acc[tm][tn], false, false);

    if (it + 1 < nIter) { storeA(cur ^ 1); storeB(cur ^ 1); }
    __syncthreads();
  }

  // ---- epilogue: bias / residual / gelu / store f32 and/or f16 ----
  #pragma unroll
  for (int tm = 0; tm < 2; ++tm) {
    #pragma unroll
    for (int tn = 0; tn < 4; ++tn) {
      const int n = blockIdx.y * TILE_N + waveN * 64 + tn * 16 + mlane;
      if (n >= N) continue;
      const float bv = bias ? bias[n] : 0.0f;
      #pragma unroll
      for (int r = 0; r < 8; ++r) {
        const int m = blockIdx.x * TILE_M + waveM * 32 + tm * 16 + hh * 8 + r;  // < M by contract
        float v = acc[tm][tn][r] + bv;
        if (res) v += res[(size_t)m * ldR + n];
        if (doGelu) v = 0.5f * v * (1.0f + erff(v * 0.70710678f));
        const size_t ci = (size_t)m * ldC + n;
        if (Cf) Cf[ci] = v;
        if (Ch) Ch[ci] = (half_t)v;
      }
    }
  }
}

// ---------------------------------------------------------------------------
// Elementwise / reshape / decomposition kernels
// ---------------------------------------------------------------------------

// streaming f32 -> f16 convert (n must be a multiple of 4)
__global__ void cvt_f16(const float* __restrict__ src, half_t* __restrict__ dst, int n)
{
  int idx = (blockIdx.x * blockDim.x + threadIdx.x) * 4;
  if (idx >= n) return;
  const float4 v = *(const float4*)(src + idx);
  v4h o; o[0] = (half_t)v.x; o[1] = (half_t)v.y; o[2] = (half_t)v.z; o[3] = (half_t)v.w;
  *(v4h*)(dst + idx) = o;
}

// moving-average decomposition of x_enc (B,512,21); writes seasonal + trend
__global__ void decomp_xenc(const float* __restrict__ x, float* __restrict__ seas,
                            float* __restrict__ trend)
{
  const int total = 128 * 512 * 21;
  int idx = blockIdx.x * blockDim.x + threadIdx.x;
  if (idx >= total) return;
  const int c = idx % 21;
  const int rest = idx / 21;
  const int l = rest % 512;
  const int b = rest / 512;
  const float* base = x + (size_t)b * 512 * 21 + c;
  float s = 0.f;
  #pragma unroll
  for (int j = -12; j <= 12; ++j) {
    int p = l + j; p = p < 0 ? 0 : (p > 511 ? 511 : p);
    s += base[(size_t)p * 21];
  }
  const float mean = s * (1.0f / 25.0f);
  trend[idx] = mean;
  seas[idx]  = x[idx] - mean;
}

// decomposition of (B,63,512) along L; writes seasonal fp32 + f16
__global__ void decomp_mid(const float* __restrict__ x, float* __restrict__ outF,
                           half_t* __restrict__ outH)
{
  const int total = 128 * 63 * 512;
  int idx = blockIdx.x * blockDim.x + threadIdx.x;
  if (idx >= total) return;
  const int d = idx & 511;
  const int rest = idx >> 9;
  const int l = rest % 63;
  const int b = rest / 63;
  const float* base = x + (size_t)b * 63 * 512 + d;
  float s = 0.f;
  #pragma unroll
  for (int j = -12; j <= 12; ++j) {
    int p = l + j; p = p < 0 ? 0 : (p > 62 ? 62 : p);
    s += base[(size_t)p * 512];
  }
  const float mean = s * (1.0f / 25.0f);
  const float v = x[idx] - mean;
  outF[idx] = v;
  outH[idx] = (half_t)v;
}

// trend_out[b,p,c] = sum_l trend[b,l,c]*trend_W[p,l] + trend_b[p]  -> d_out
__global__ void trend_head(const float* __restrict__ trend, const float* __restrict__ tW,
                           const float* __restrict__ tb, float* __restrict__ out)
{
  const int total = 128 * 96 * 21;
  int idx = blockIdx.x * blockDim.x + threadIdx.x;
  if (idx >= total) return;
  const int c = idx % 21;
  const int p = (idx / 21) % 96;
  const int b = idx / (96 * 21);
  const float* tr = trend + (size_t)b * 512 * 21 + c;
  const float* wp = tW + (size_t)p * 512;
  float s = tb[p];
  for (int l = 0; l < 512; ++l) s += wp[l] * tr[(size_t)l * 21];
  out[idx] = s;
}

// im2col patches: xp[(b*63+n)*336 + c*16+t] = seasonal[b, 8n+t, c]
__global__ void patch_gather(const float* __restrict__ seas, half_t* __restrict__ xp)
{
  const int total = 128 * 63 * 336;
  int idx = blockIdx.x * blockDim.x + threadIdx.x;
  if (idx >= total) return;
  const int q = idx % 336;
  const int n = (idx / 336) % 63;
  const int b = idx / (336 * 63);
  const int c = q >> 4;
  const int t = q & 15;
  xp[idx] = (half_t)seas[((size_t)b * 512 + 8 * n + t) * 21 + c];
}

// q16 (B,63,512) -> qT (B*512, 64)  [col 63 zero-padded]
__global__ void q_to_qT(const half_t* __restrict__ q, half_t* __restrict__ qT)
{
  const int total = 128 * 512 * 64;
  int idx = blockIdx.x * blockDim.x + threadIdx.x;
  if (idx >= total) return;
  const int l = idx & 63;
  const int d = (idx >> 6) & 511;
  const int b = idx >> 15;
  qT[idx] = (l < 63) ? q[((size_t)b * 63 + l) * 512 + d] : (half_t)0.f;
}

// forward DFT basis, N-major (64 m-rows x 64 l-cols, K=63 used):
// rows 0..30 = cos, rows 32..62 = -sin, rows 31/63 = 0
__global__ void dft_basis(half_t* __restrict__ Bd)
{
  int idx = blockIdx.x * blockDim.x + threadIdx.x;
  if (idx >= 64 * 64) return;
  const int l = idx & 63, m = idx >> 6;
  const float w = 6.2831853071795864769f / 63.0f;
  float v = 0.f;
  if (l < 63) {
    if (m < 31) v = __cosf(w * (float)(l * m));
    else if (m >= 32 && m < 63) v = -__sinf(w * (float)(l * (m - 32)));
  }
  Bd[idx] = (half_t)v;
}

// inverse DFT basis, N-major (64 l-rows x 64 k-cols):
// (pocketfft irfft: Im(X0) ignored, modes 1..30 doubled, 1/63 norm)
__global__ void idft_basis(half_t* __restrict__ Bi)
{
  int idx = blockIdx.x * blockDim.x + threadIdx.x;
  if (idx >= 64 * 64) return;
  const int k = idx & 63, l = idx >> 6;
  const float w = 6.2831853071795864769f / 63.0f;
  float v = 0.f;
  if (l < 63) {
    if (k == 0) v = 1.0f / 63.0f;
    else if (k < 31) v = (2.0f / 63.0f) * __cosf(w * (float)(k * l));
    else if (k >= 32 && k < 63) v = -(2.0f / 63.0f) * __sinf(w * (float)((k - 32) * l));
  }
  Bi[idx] = (half_t)v;
}

// gather DFT output into per-(h,m) mix operand A: (248, 128b, 128=[Xr|Xi])
__global__ void mix_gather(const half_t* __restrict__ Xd, half_t* __restrict__ Xm)
{
  const int total = 248 * 128 * 128;
  int idx = blockIdx.x * blockDim.x + threadIdx.x;
  if (idx >= total) return;
  const int col = idx & 127;
  const int b   = (idx >> 7) & 127;
  const int hm  = idx >> 14;
  const int h = hm / 31, m = hm % 31;
  const int e = col & 63, im = col >> 6;
  Xm[idx] = Xd[((size_t)b * 512 + h * 64 + e) * 64 + (im ? 32 + m : m)];
}

// build complex mixing matrix per (h,m) in N-major layout:
// Bm[hm][n=(f,rf)][k=(e,re)] = [[Wr,Wi],[-Wi,Wr]] transposed to NxK
__global__ void bmix_build(const float* __restrict__ wr, const float* __restrict__ wi,
                           half_t* __restrict__ Bm)
{
  const int total = 248 * 128 * 128;
  int idx = blockIdx.x * blockDim.x + threadIdx.x;
  if (idx >= total) return;
  const int kd  = idx & 127;          // K index (input): e, re
  const int n   = (idx >> 7) & 127;   // N index (output): f, rf
  const int hm  = idx >> 14;
  const int h = hm / 31, m = hm % 31;
  const int e = kd & 63, re = kd >> 6;
  const int f = n & 63,  rf = n >> 6;
  const size_t wix = ((((size_t)h * 64 + e) * 64 + f) * 31) + m;
  float v;
  if (!re && !rf)      v =  wr[wix];
  else if (!re &&  rf) v =  wi[wix];
  else if ( re && !rf) v = -wi[wix];
  else                 v =  wr[wix];
  Bm[idx] = (half_t)v;
}

// rearrange mixed spectrum for iDFT: YT[(b*512+h*64+f), 64=[Yr(32)|Yi(32)]]
__global__ void y_gather(const half_t* __restrict__ Ym, half_t* __restrict__ YT)
{
  const int total = 128 * 512 * 64;
  int idx = blockIdx.x * blockDim.x + threadIdx.x;
  if (idx >= total) return;
  const int col = idx & 63;
  const int d   = (idx >> 6) & 511;
  const int b   = idx >> 15;
  const int h = d >> 6, f = d & 63;
  half_t v = (half_t)0.f;
  if (col < 31)                 v = Ym[(size_t)(h * 31 + col) * 16384 + b * 128 + f];
  else if (col >= 32 && col < 63) v = Ym[(size_t)(h * 31 + (col - 32)) * 16384 + b * 128 + 64 + f];
  YT[idx] = v;
}

// reference does y.reshape(B, L, H*E) on (B,H,F,L) -- a RAW reshape.
// attn[(b, flat)] with flat=l'*512+d' maps to (h = flat/4032, f = (flat%4032)/63, l = flat%63)
__global__ void attn_scatter(const half_t* __restrict__ yt, half_t* __restrict__ attn)
{
  const int total = 128 * 63 * 512;
  int idx = blockIdx.x * blockDim.x + threadIdx.x;
  if (idx >= total) return;
  const int flat = idx % 32256;
  const int b    = idx / 32256;
  const int h  = flat / 4032;
  const int r  = flat % 4032;
  const int f  = r / 63;
  const int lt = r % 63;
  attn[idx] = yt[((size_t)b * 512 + h * 64 + f) * 64 + lt];
}

// layernorm over last dim (512), wave-per-row
__global__ __launch_bounds__(256)
void layernorm_rows(const float* __restrict__ x, const float* __restrict__ g,
                    const float* __restrict__ be, float* __restrict__ out)
{
  const int w = threadIdx.x >> 5;
  const int lane = threadIdx.x & 31;
  const int row = blockIdx.x * 8 + w;
  if (row >= 8064) return;
  const float* xr = x + (size_t)row * 512;
  float s = 0.f, ss = 0.f;
  for (int j = lane; j < 512; j += 32) { const float v = xr[j]; s += v; ss += v * v; }
  #pragma unroll
  for (int o = 16; o > 0; o >>= 1) { s += __shfl_xor(s, o, 32); ss += __shfl_xor(ss, o, 32); }
  const float mu = s * (1.0f / 512.0f);
  const float var = ss * (1.0f / 512.0f) - mu * mu;
  const float inv = rsqrtf(var + 1e-5f);
  for (int j = lane; j < 512; j += 32)
    out[(size_t)row * 512 + j] = (xr[j] - mu) * inv * g[j] + be[j];
}

// subtract per-(b,d) mean over L=63; emit f16 for the head GEMM
__global__ void sub_seqmean(const float* __restrict__ xh, half_t* __restrict__ out)
{
  const int total = 128 * 512;
  int idx = blockIdx.x * blockDim.x + threadIdx.x;
  if (idx >= total) return;
  const int d = idx & 511;
  const int b = idx >> 9;
  const float* p = xh + (size_t)b * 63 * 512 + d;
  float s = 0.f;
  for (int l = 0; l < 63; ++l) s += p[(size_t)l * 512];
  const float mu = s * (1.0f / 63.0f);
  for (int l = 0; l < 63; ++l)
    out[((size_t)b * 63 + l) * 512 + d] = (half_t)(p[(size_t)l * 512] - mu);
}

// ---------------------------------------------------------------------------
extern "C" void kernel_launch(void* const* d_in, const int* in_sizes, int n_in,
                              void* d_out, int out_size, void* d_ws, size_t ws_size,
                              hipStream_t stream)
{
  (void)in_sizes; (void)n_in; (void)out_size; (void)ws_size;
  const float* x_enc   = (const float*)d_in[0];
  const float* patch_W = (const float*)d_in[4];
  const float* patch_b = (const float*)d_in[5];
  const float* Wq      = (const float*)d_in[6];
  const float* bq      = (const float*)d_in[7];
  const float* Wo      = (const float*)d_in[8];
  const float* bo      = (const float*)d_in[9];
  const float* four_wr = (const float*)d_in[10];
  const float* four_wi = (const float*)d_in[11];
  const float* conv1_W = (const float*)d_in[12];
  const float* conv2_W = (const float*)d_in[13];
  const float* gamma   = (const float*)d_in[14];
  const float* beta    = (const float*)d_in[15];
  const float* head_W  = (const float*)d_in[16];
  const float* head_b  = (const float*)d_in[17];
  const float* trend_W = (const float*)d_in[18];
  const float* trend_b = (const float*)d_in[19];
  float* out = (float*)d_out;

  char* wsb = (char*)d_ws;
  size_t off = 0;
  auto alloc = [&](size_t bytes) -> void* {
    void* p = wsb + off;
    off += (bytes + 255) & ~(size_t)255;
    return p;
  };

  float*  seas   = (float*) alloc((size_t)128 * 512 * 21 * 4);
  float*  trendb = (float*) alloc((size_t)128 * 512 * 21 * 4);
  half_t* xp16   = (half_t*)alloc((size_t)8064 * 336 * 2);
  float*  enc    = (float*) alloc((size_t)8064 * 512 * 4);
  half_t* enc16  = (half_t*)alloc((size_t)8064 * 512 * 2);
  half_t* q16    = (half_t*)alloc((size_t)8064 * 512 * 2);
  half_t* qT16   = (half_t*)alloc((size_t)65536 * 64 * 2);
  half_t* Xdft16 = (half_t*)alloc((size_t)65536 * 64 * 2);
  half_t* Xmix16 = (half_t*)alloc((size_t)248 * 16384 * 2);
  half_t* Ymix16 = (half_t*)alloc((size_t)248 * 16384 * 2);
  half_t* YT16   = (half_t*)alloc((size_t)65536 * 64 * 2);
  half_t* yt16   = (half_t*)alloc((size_t)65536 * 64 * 2);
  half_t* attn16 = (half_t*)alloc((size_t)8064 * 512 * 2);
  float*  sum1   = (float*) alloc((size_t)8064 * 512 * 4);
  float*  xbuf   = (float*) alloc((size_t)8064 * 512 * 4);
  half_t* x16    = (half_t*)alloc((size_t)8064 * 512 * 2);
  half_t* y116   = (half_t*)alloc((size_t)8064 * 2048 * 2);
  float*  sum2   = (float*) alloc((size_t)8064 * 512 * 4);
  float*  xh     = (float*) alloc((size_t)8064 * 512 * 4);
  half_t* encf16 = (half_t*)alloc((size_t)8064 * 512 * 2);
  // weight-native f16 copies (N x K, no transpose needed)
  half_t* patch16 = (half_t*)alloc((size_t)512 * 336 * 2);
  half_t* Wq16    = (half_t*)alloc((size_t)2 * 512 * 512 * 2);
  half_t* Wo16    = (half_t*)alloc((size_t)2 * 512 * 512 * 2);
  half_t* c116    = (half_t*)alloc((size_t)2 * 2048 * 512 * 2);
  half_t* c216    = (half_t*)alloc((size_t)2 * 512 * 2048 * 2);
  half_t* head16  = (half_t*)alloc((size_t)2016 * 32256 * 2);
  half_t* Bdft    = (half_t*)alloc((size_t)64 * 64 * 2);
  half_t* Bidft   = (half_t*)alloc((size_t)64 * 64 * 2);
  half_t* Bmix    = (half_t*)alloc((size_t)2 * 248 * 16384 * 2);

  auto cdiv = [](int a, int b) { return (a + b - 1) / b; };
  const int T = 256;

  // ---- weight prep: pure streaming converts (deterministic each call) ----
  cvt_f16<<<cdiv(512 * 336,        T * 4), T, 0, stream>>>(patch_W, patch16, 512 * 336);
  cvt_f16<<<cdiv(2 * 512 * 512,    T * 4), T, 0, stream>>>(Wq,      Wq16,    2 * 512 * 512);
  cvt_f16<<<cdiv(2 * 512 * 512,    T * 4), T, 0, stream>>>(Wo,      Wo16,    2 * 512 * 512);
  cvt_f16<<<cdiv(2 * 2048 * 512,   T * 4), T, 0, stream>>>(conv1_W, c116,    2 * 2048 * 512);
  cvt_f16<<<cdiv(2 * 512 * 2048,   T * 4), T, 0, stream>>>(conv2_W, c216,    2 * 512 * 2048);
  cvt_f16<<<cdiv(2016 * 32256,     T * 4), T, 0, stream>>>(head_W,  head16,  2016 * 32256);
  for (int l = 0; l < 2; ++l)
    bmix_build<<<cdiv(248 * 16384, T), T, 0, stream>>>(four_wr + (size_t)l * 1015808,
                                                       four_wi + (size_t)l * 1015808,
                                                       Bmix + (size_t)l * 248 * 16384);
  dft_basis <<<cdiv(64 * 64, T), T, 0, stream>>>(Bdft);
  idft_basis<<<cdiv(64 * 64, T), T, 0, stream>>>(Bidft);

  // ---- decomposition + trend branch (writes d_out fully) ----
  decomp_xenc<<<cdiv(128 * 512 * 21, T), T, 0, stream>>>(x_enc, seas, trendb);
  trend_head <<<cdiv(128 * 96 * 21, T),  T, 0, stream>>>(trendb, trend_W, trend_b, out);

  // ---- patch embedding ----
  patch_gather<<<cdiv(128 * 63 * 336, T), T, 0, stream>>>(seas, xp16);
  gemm_f16_wmma<<<dim3(63, 4, 1), T, 0, stream>>>(xp16, 336, 0, patch16, 336, 0,
      enc, enc16, 512, 0, patch_b, nullptr, 0, 8064, 512, 336, 0);

  // ---- encoder layers ----
  for (int l = 0; l < 2; ++l) {
    const half_t* Wq16_l = Wq16 + (size_t)l * 262144;
    const half_t* Wo16_l = Wo16 + (size_t)l * 262144;
    const half_t* c116_l = c116 + (size_t)l * 1048576;
    const half_t* c216_l = c216 + (size_t)l * 1048576;
    const half_t* Bmix_l = Bmix + (size_t)l * 248 * 16384;
    const float*  bq_l = bq + (size_t)l * 512;
    const float*  bo_l = bo + (size_t)l * 512;

    // q projection
    gemm_f16_wmma<<<dim3(63, 4, 1), T, 0, stream>>>(enc16, 512, 0, Wq16_l, 512, 0,
        nullptr, q16, 512, 0, bq_l, nullptr, 0, 8064, 512, 512, 0);
    q_to_qT<<<cdiv(128 * 512 * 64, T), T, 0, stream>>>(q16, qT16);

    // forward DFT (length-63 rfft as GEMM vs cos/sin basis)
    gemm_f16_wmma<<<dim3(512, 1, 1), T, 0, stream>>>(qT16, 64, 0, Bdft, 64, 0,
        nullptr, Xdft16, 64, 0, nullptr, nullptr, 0, 65536, 64, 63, 0);

    // per-(head,mode) complex mixing as 248 batched real 128x128x128 GEMMs
    mix_gather<<<cdiv(248 * 16384, T), T, 0, stream>>>(Xdft16, Xmix16);
    gemm_f16_wmma<<<dim3(1, 1, 248), T, 0, stream>>>(Xmix16, 128, 16384, Bmix_l, 128, 16384,
        nullptr, Ymix16, 128, 16384, nullptr, nullptr, 0, 128, 128, 128, 0);

    // inverse DFT
    y_gather<<<cdiv(128 * 512 * 64, T), T, 0, stream>>>(Ymix16, YT16);
    gemm_f16_wmma<<<dim3(512, 1, 1), T, 0, stream>>>(YT16, 64, 0, Bidft, 64, 0,
        nullptr, yt16, 64, 0, nullptr, nullptr, 0, 65536, 64, 64, 0);
    attn_scatter<<<cdiv(128 * 63 * 512, T), T, 0, stream>>>(yt16, attn16);

    // output projection + residual, then seasonal decomposition
    gemm_f16_wmma<<<dim3(63, 4, 1), T, 0, stream>>>(attn16, 512, 0, Wo16_l, 512, 0,
        sum1, nullptr, 512, 0, bo_l, enc, 512, 8064, 512, 512, 0);
    decomp_mid<<<cdiv(128 * 63 * 512, T), T, 0, stream>>>(sum1, xbuf, x16);

    // FFN: gelu(x @ c1^T) @ c2^T + x, then decomposition -> next enc
    gemm_f16_wmma<<<dim3(63, 16, 1), T, 0, stream>>>(x16, 512, 0, c116_l, 512, 0,
        nullptr, y116, 2048, 0, nullptr, nullptr, 0, 8064, 2048, 512, 1);
    gemm_f16_wmma<<<dim3(63, 4, 1), T, 0, stream>>>(y116, 2048, 0, c216_l, 2048, 0,
        sum2, nullptr, 512, 0, nullptr, xbuf, 512, 8064, 512, 2048, 0);
    decomp_mid<<<cdiv(128 * 63 * 512, T), T, 0, stream>>>(sum2, enc, enc16);
  }

  // ---- final norm + head (accumulates onto trend_out already in d_out) ----
  layernorm_rows<<<1008, T, 0, stream>>>(enc, gamma, beta, xh);
  sub_seqmean<<<cdiv(128 * 512, T), T, 0, stream>>>(xh, encf16);
  gemm_f16_wmma<<<dim3(1, 16, 1), T, 0, stream>>>(encf16, 32256, 0, head16, 32256, 0,
      out, nullptr, 2016, 0, head_b, out, 2016, 128, 2016, 32256, 0);
}
